// FeatureExtractor_force_88038239634144
// MI455X (gfx1250) — matrix operand order repaired and checked by reading, not compiled, not run on hardware
//
#include <hip/hip_runtime.h>
#include <hip/hip_bf16.h>
#include <stdint.h>

// ---------------------------------------------------------------------------
// Types / helpers for CDNA5 WMMA (wave32, 16x16x32 bf16 -> f32)
// ---------------------------------------------------------------------------
typedef __bf16 bhalf;
typedef __attribute__((ext_vector_type(16))) __bf16 v16bf;
typedef __attribute__((ext_vector_type(8)))  float  v8f;

#define DEVFN static __device__ __forceinline__

constexpr float BN_SCALE = 0.99999500003749977f; // 1/sqrt(1+1e-5)
constexpr int BATCH = 16;
constexpr int NPTS0 = 16896;   // 16384 partial + 512 kp
constexpr int S1 = 512;
constexpr int S2 = 128;

DEVFN bhalf f2b(float x) { return (bhalf)x; }

// A-operand fragment, row-major [16][K] tile at (row0, k0), ld = row stride.
// ISA 7.12.2 16-bit A 16x32: lanes 0-15 K={0..7,16..23}, lanes 16-31 K={8..15,24..31}
DEVFN v16bf frag_a_rm(const bhalf* base, int ld, int k0, int lane) {
  const int m  = lane & 15;
  const int kh = (lane >> 4) << 3;
  const bhalf* p = base + (size_t)m * ld + (k0 + kh);
  union { v16bf v; uint4 q[2]; } f;
  f.q[0] = *reinterpret_cast<const uint4*>(p);       // K +0..7
  f.q[1] = *reinterpret_cast<const uint4*>(p + 16);  // K +16..23
  return f.v;
}

// B-operand fragment from row-major weights W[N][Kp] tile at (n0, k0).
// B layout: lanes 0-15 hold K={0..7,16..23}, lanes 16-31 hold K={8..15,24..31} (2 per VGPR).
DEVFN v16bf frag_b_rm(const bhalf* base, int ld, int k0, int lane) {
  const int n  = lane & 15;
  const int kh = (lane >> 4) << 3;
  const bhalf* p = base + (size_t)n * ld + (k0 + kh);
  union { v16bf v; uint4 q[2]; } f;
  f.q[0] = *reinterpret_cast<const uint4*>(p);       // K +0..7
  f.q[1] = *reinterpret_cast<const uint4*>(p + 16);  // K +16..23
  return f.v;
}

DEVFN v8f wmma_bf16(v16bf a, v16bf b, v8f c) {
  return __builtin_amdgcn_wmma_f32_16x16x32_bf16(false, a, false, b, (short)0, c,
                                                 false, false);
}

// ---------------------------------------------------------------------------
// Weight convert: f32 [O][K] -> bf16 [O][Kp] zero-padded row-major
// ---------------------------------------------------------------------------
__global__ void __launch_bounds__(256) convert_pad_w(const float* __restrict__ w,
                                                     bhalf* __restrict__ o,
                                                     int K, int Kp, int total) {
  int t = blockIdx.x * blockDim.x + threadIdx.x;
  if (t >= total) return;
  int r = t / Kp, c = t - r * Kp;
  o[t] = f2b((c < K) ? w[(size_t)r * K + c] : 0.0f);
}

// ---------------------------------------------------------------------------
// Concat partial_cloud (B,3,16384) + kp (B,3,512) -> point-major [B*N][3]
// ---------------------------------------------------------------------------
__global__ void __launch_bounds__(256) concat_xyz(const float* __restrict__ pc,
                                                  const float* __restrict__ kp,
                                                  float* __restrict__ pm) {
  int t = blockIdx.x * blockDim.x + threadIdx.x;
  if (t >= BATCH * NPTS0) return;
  int b = t / NPTS0, i = t - b * NPTS0;
  float x, y, z;
  if (i < 16384) {
    const float* s = pc + (size_t)b * 3 * 16384;
    x = s[i]; y = s[16384 + i]; z = s[2 * 16384 + i];
  } else {
    int j = i - 16384;
    const float* s = kp + (size_t)b * 3 * 512;
    x = s[j]; y = s[512 + j]; z = s[2 * 512 + j];
  }
  float* o = pm + (size_t)t * 3;
  o[0] = x; o[1] = y; o[2] = z;
}

// ---------------------------------------------------------------------------
// Farthest point sampling: one workgroup per batch, dist[] in dynamic LDS
// ---------------------------------------------------------------------------
__global__ void __launch_bounds__(1024) fps_kernel(const float* __restrict__ xyz,
                                                   int N, int S,
                                                   int* __restrict__ out_idx) {
  extern __shared__ float fps_sm[];
  float* dist = fps_sm;                    // [N]
  float* rv   = fps_sm + N;                // [blockDim]
  int*   ri   = (int*)(rv + blockDim.x);   // [blockDim]
  __shared__ float cur[3];
  const int b = blockIdx.x;
  const float* X = xyz + (size_t)b * N * 3;
  for (int i = threadIdx.x; i < N; i += blockDim.x) dist[i] = 1e10f;
  if (threadIdx.x == 0) {
    cur[0] = X[0]; cur[1] = X[1]; cur[2] = X[2];
    out_idx[(size_t)b * S] = 0;
  }
  __syncthreads();
  for (int s = 1; s < S; ++s) {
    const float px = cur[0], py = cur[1], pz = cur[2];
    float bmax = -1.0f; int bidx = 0;
    for (int i = threadIdx.x; i < N; i += blockDim.x) {
      float dx = X[3 * i] - px, dy = X[3 * i + 1] - py, dz = X[3 * i + 2] - pz;
      float d = dx * dx + dy * dy + dz * dz;
      float dd = fminf(dist[i], d);
      dist[i] = dd;
      if (dd > bmax) { bmax = dd; bidx = i; }
    }
    rv[threadIdx.x] = bmax; ri[threadIdx.x] = bidx;
    __syncthreads();
    for (int off = blockDim.x >> 1; off > 0; off >>= 1) {
      if (threadIdx.x < off && rv[threadIdx.x + off] > rv[threadIdx.x]) {
        rv[threadIdx.x] = rv[threadIdx.x + off];
        ri[threadIdx.x] = ri[threadIdx.x + off];
      }
      __syncthreads();
    }
    if (threadIdx.x == 0) {
      int w = ri[0];
      cur[0] = X[3 * w]; cur[1] = X[3 * w + 1]; cur[2] = X[3 * w + 2];
      out_idx[(size_t)b * S + s] = w;
    }
    __syncthreads();
  }
}

// Gather sampled centroids -> point-major and channel-major copies
__global__ void __launch_bounds__(256) gather_xyz(const float* __restrict__ pm,
                                                  const int* __restrict__ fidx,
                                                  float* __restrict__ nxpm,
                                                  float* __restrict__ nxcm,
                                                  int N, int S) {
  int t = blockIdx.x * blockDim.x + threadIdx.x;
  if (t >= BATCH * S) return;
  int b = t / S, s = t - b * S;
  int id = fidx[t];
  const float* p = pm + ((size_t)b * N + id) * 3;
  float x = p[0], y = p[1], z = p[2];
  float* o = nxpm + (size_t)t * 3;
  o[0] = x; o[1] = y; o[2] = z;
  nxcm[((size_t)b * 3 + 0) * S + s] = x;
  nxcm[((size_t)b * 3 + 1) * S + s] = y;
  nxcm[((size_t)b * 3 + 2) * S + s] = z;
}

// ---------------------------------------------------------------------------
// KNN: thread-per-query, LDS point tiles, unrolled predicated register top-K
// ---------------------------------------------------------------------------
template<int K>
__global__ void __launch_bounds__(128) knn_kernel(const float* __restrict__ Q,
                                                  const float* __restrict__ P,
                                                  int S, int N,
                                                  int* __restrict__ out) {
  constexpr int TILE = 256;
  __shared__ float tp[TILE * 3];
  const int b = blockIdx.y;
  const int q = blockIdx.x * blockDim.x + threadIdx.x;
  const bool valid = q < S;
  float qx = 0.f, qy = 0.f, qz = 0.f;
  if (valid) {
    const float* qp = Q + ((size_t)b * S + q) * 3;
    qx = qp[0]; qy = qp[1]; qz = qp[2];
  }
  float bd[K]; int bi[K];
#pragma unroll
  for (int t = 0; t < K; ++t) { bd[t] = 3e38f; bi[t] = 0; }
  for (int base = 0; base < N; base += TILE) {
    const int cnt = min(TILE, N - base);
    __syncthreads();
    for (int i = threadIdx.x; i < cnt * 3; i += blockDim.x)
      tp[i] = P[((size_t)b * N + base) * 3 + i];
    __syncthreads();
    if (valid) {
      for (int i = 0; i < cnt; ++i) {
        float dx = tp[3 * i] - qx, dy = tp[3 * i + 1] - qy, dz = tp[3 * i + 2] - qz;
        float d = dx * dx + dy * dy + dz * dz;
        if (d < bd[K - 1]) {
          const int pid = base + i;
#pragma unroll
          for (int j = K - 1; j >= 1; --j) {
            if (bd[j - 1] > d)      { bd[j] = bd[j - 1]; bi[j] = bi[j - 1]; }
            else if (bd[j] > d)     { bd[j] = d;         bi[j] = pid;       }
          }
          if (bd[0] > d) { bd[0] = d; bi[0] = pid; }
        }
      }
    }
  }
  if (valid) {
    int* o = out + ((size_t)b * S + q) * K;
#pragma unroll
    for (int t = 0; t < K; ++t) o[t] = bi[t];
  }
}

// ---------------------------------------------------------------------------
// SA module: gather 16 centroids x 16 neighbors into LDS, 2-layer WMMA MLP,
// fused max over neighbors (M-tile == one centroid's 16 neighbors).
// ---------------------------------------------------------------------------
template<int CF, int CINP, int C1, int C2>
__global__ void __launch_bounds__(256) sa_grouped_mlp(
    const float* __restrict__ xyz, const float* __restrict__ feats,
    const float* __restrict__ nx, const int* __restrict__ idx,
    const bhalf* __restrict__ W1, const float* __restrict__ b1,
    const bhalf* __restrict__ W2, const float* __restrict__ b2,
    float* __restrict__ outF, bhalf* __restrict__ outB, int N, int S) {
  extern __shared__ bhalf sa_sm[];
  bhalf* sA = sa_sm;                   // [256][CINP]
  bhalf* sH = sa_sm + 256 * CINP;      // [256][C1]
  const int b = blockIdx.y;
  const int s0 = blockIdx.x * 16;
  const int t = threadIdx.x;
  {
    const int p = t >> 4, j = t & 15;
    const int s = s0 + p;
    const int id = idx[((size_t)b * S + s) * 16 + j];
    const float* cp = nx + ((size_t)b * S + s) * 3;
    const float* pp = xyz + ((size_t)b * N + id) * 3;
    bhalf* row = sA + (size_t)t * CINP;
    row[0] = f2b(pp[0] - cp[0]);
    row[1] = f2b(pp[1] - cp[1]);
    row[2] = f2b(pp[2] - cp[2]);
    const float* fp = feats + ((size_t)b * N + id) * CF;
#pragma unroll 4
    for (int c = 0; c < CF; ++c) row[3 + c] = f2b(fp[c]);
    for (int c = 3 + CF; c < CINP; ++c) row[c] = f2b(0.0f);
  }
  __syncthreads();
  const int wave = t >> 5, lane = t & 31;
  const int nl = lane & 15, mh = (lane >> 4) << 3;
  // layer 1: bias + relu -> bf16 LDS
  constexpr int NT1 = C1 / 16;
  for (int tile = wave; tile < 16 * NT1; tile += 8) {
    const int mt = tile / NT1, nt = tile % NT1;
    v8f acc = {};
#pragma unroll
    for (int k0 = 0; k0 < CINP; k0 += 32) {
      v16bf a = frag_a_rm(sA + (size_t)mt * 16 * CINP, CINP, k0, lane);
      v16bf w = frag_b_rm(W1 + (size_t)(nt * 16) * CINP, CINP, k0, lane);
      acc = wmma_bf16(a, w, acc);
    }
    const int col = nt * 16 + nl;
    const float bb = b1[col];
#pragma unroll
    for (int r = 0; r < 8; ++r)
      sH[(size_t)(mt * 16 + mh + r) * C1 + col] = f2b(fmaxf(acc[r] + bb, 0.0f));
  }
  __syncthreads();
  // layer 2: WMMA + in-register max over the 16 neighbors, bias after max
  constexpr int NT2 = C2 / 16;
  for (int tile = wave; tile < 16 * NT2; tile += 8) {
    const int mt = tile / NT2, nt = tile % NT2;
    v8f acc = {};
#pragma unroll
    for (int k0 = 0; k0 < C1; k0 += 32) {
      v16bf a = frag_a_rm(sH + (size_t)mt * 16 * C1, C1, k0, lane);
      v16bf w = frag_b_rm(W2 + (size_t)(nt * 16) * C1, C1, k0, lane);
      acc = wmma_bf16(a, w, acc);
    }
    float mx = acc[0];
#pragma unroll
    for (int r = 1; r < 8; ++r) mx = fmaxf(mx, acc[r]);
    mx = fmaxf(mx, __shfl_xor(mx, 16));          // fold lane halves (wave32)
    if (lane < 16) {
      const int col = nt * 16 + lane;
      const float v = mx + b2[col];
      const size_t o = ((size_t)b * S + s0 + mt) * C2 + col;
      outF[o] = v;
      if (outB) outB[o] = f2b(v);
    }
  }
}

// ---------------------------------------------------------------------------
// Generic row-major bf16 WMMA GEMM.
//   - weight tile (16 x Kp) staged once per block into LDS via CDNA5
//     async-to-LDS DMA (ASYNCcnt), then reused by all 8 waves as ds_load_b128
//   - each wave owns TWO 16-row M-tiles (two independent WMMA chains)
//   - grid: (rows/256, N/16); dynamic LDS: 32*Kp bytes
// flags: 1=relu, 2=bn(gamma,beta around bias), 4=+residual
// ---------------------------------------------------------------------------
__global__ void __launch_bounds__(256) gemm_rm(
    const bhalf* __restrict__ A, const bhalf* __restrict__ W,
    const float* __restrict__ bias, const float* __restrict__ gamma,
    const float* __restrict__ beta, const float* __restrict__ resid,
    float* __restrict__ outF, bhalf* __restrict__ outB,
    int Kp, int N, int flags) {
  extern __shared__ bhalf gw_sm[];     // [16][Kp] weight tile
  const int n0 = blockIdx.y * 16;
  // ---- async-stage the weight tile: 16*Kp bf16 = 2*Kp 16-byte chunks ----
  {
    const int chunks = Kp * 2;
    const unsigned lds_base = (unsigned)(uintptr_t)gw_sm;
    const bhalf* wsrc = W + (size_t)n0 * Kp;
    for (int c = threadIdx.x; c < chunks; c += blockDim.x) {
      unsigned lds_addr = lds_base + (unsigned)c * 16u;
      const bhalf* src = wsrc + (size_t)c * 8;
      asm volatile("global_load_async_to_lds_b128 %0, %1, off"
                   :: "v"(lds_addr), "v"(src) : "memory");
    }
    asm volatile("s_wait_asynccnt 0" ::: "memory");
    __syncthreads();
  }
  const int lane = threadIdx.x & 31;
  const int wave = threadIdx.x >> 5;
  const int row0 = blockIdx.x * 256 + wave * 32;
  const bhalf* Abase0 = A + (size_t)row0 * Kp;
  const bhalf* Abase1 = Abase0 + (size_t)16 * Kp;
  v8f acc0 = {};
  v8f acc1 = {};
  for (int k0 = 0; k0 < Kp; k0 += 32) {
    v16bf w  = frag_b_rm(gw_sm, Kp, k0, lane);
    v16bf a0 = frag_a_rm(Abase0, Kp, k0, lane);
    v16bf a1 = frag_a_rm(Abase1, Kp, k0, lane);
    acc0 = wmma_bf16(a0, w, acc0);
    acc1 = wmma_bf16(a1, w, acc1);
  }
  const int nl = lane & 15;
  const int col = n0 + nl;
  const int mh = (lane >> 4) << 3;
  const float bs = bias ? bias[col] : 0.0f;
  float g = 1.0f, bt = 0.0f;
  if (flags & 2) { g = gamma[col] * BN_SCALE; bt = beta[col]; }
#pragma unroll
  for (int half = 0; half < 2; ++half) {
    const v8f acc = half ? acc1 : acc0;
    const int rb = row0 + half * 16 + mh;
#pragma unroll
    for (int r = 0; r < 8; ++r) {
      float v = acc[r] + bs;
      if (flags & 2) v = g * v + bt;
      if (flags & 1) v = fmaxf(v, 0.0f);
      const size_t o = (size_t)(rb + r) * N + col;
      if (flags & 4) v += resid[o];
      if (outF) outF[o] = v;
      if (outB) outB[o] = f2b(v);
    }
  }
}

// ---------------------------------------------------------------------------
// Transformer helpers
// ---------------------------------------------------------------------------
// pos-MLP layer 1: h = relu(bn(pos1(pos_rel)))  (K=3 -> plain VALU)
__global__ void __launch_bounds__(256) pos_mlp1(
    const float* __restrict__ pos, const int* __restrict__ idx,
    const float* __restrict__ w1, const float* __restrict__ b1,
    const float* __restrict__ g, const float* __restrict__ bb,
    bhalf* __restrict__ hout, int n, int k, int total) {
  __shared__ float sw[192], sb[64], sg[64], sbb[64];
  for (int i = threadIdx.x; i < 192; i += blockDim.x) sw[i] = w1[i];
  for (int i = threadIdx.x; i < 64; i += blockDim.x) {
    sb[i] = b1[i]; sg[i] = g[i]; sbb[i] = bb[i];
  }
  __syncthreads();
  const int pair = blockIdx.x * blockDim.x + threadIdx.x;
  if (pair >= total) return;
  const int bi = pair / (n * k);
  const int rem = pair - bi * n * k;
  const int i = rem / k;
  const int nb = idx[pair];
  const float* pi = pos + ((size_t)bi * n + i) * 3;
  const float* pj = pos + ((size_t)bi * n + nb) * 3;
  const float rx = pi[0] - pj[0], ry = pi[1] - pj[1], rz = pi[2] - pj[2];
  bhalf* o = hout + (size_t)pair * 64;
#pragma unroll 8
  for (int c = 0; c < 64; ++c) {
    float h = sw[c * 3] * rx + sw[c * 3 + 1] * ry + sw[c * 3 + 2] * rz + sb[c];
    h = sg[c] * (h * BN_SCALE) + sbb[c];
    o[c] = f2b(fmaxf(h, 0.0f));
  }
}

// attn_in = q[i] - k[idx] + pe (bf16); value = v[idx] + pe (f32)
__global__ void __launch_bounds__(256) build_attn_value(
    const float* __restrict__ q, const float* __restrict__ kf,
    const float* __restrict__ vf, const float* __restrict__ pe,
    const int* __restrict__ idx, bhalf* __restrict__ ain,
    float* __restrict__ val, int n, int k, int total) {
  const int pair = blockIdx.x * blockDim.x + threadIdx.x;
  if (pair >= total) return;
  const int bi = pair / (n * k);
  const int rem = pair - bi * n * k;
  const int i = rem / k;
  const int nb = idx[pair];
  const float4* qr = (const float4*)(q + ((size_t)bi * n + i) * 64);
  const float4* kr = (const float4*)(kf + ((size_t)bi * n + nb) * 64);
  const float4* vr = (const float4*)(vf + ((size_t)bi * n + nb) * 64);
  const float4* pr = (const float4*)(pe + (size_t)pair * 64);
  bhalf* ao = ain + (size_t)pair * 64;
  float4* vo = (float4*)(val + (size_t)pair * 64);
#pragma unroll
  for (int c4 = 0; c4 < 16; ++c4) {
    const float4 qv = qr[c4], kv = kr[c4], vv = vr[c4], pv = pr[c4];
    ao[c4 * 4 + 0] = f2b(qv.x - kv.x + pv.x);
    ao[c4 * 4 + 1] = f2b(qv.y - kv.y + pv.y);
    ao[c4 * 4 + 2] = f2b(qv.z - kv.z + pv.z);
    ao[c4 * 4 + 3] = f2b(qv.w - kv.w + pv.w);
    float4 ov;
    ov.x = vv.x + pv.x; ov.y = vv.y + pv.y;
    ov.z = vv.z + pv.z; ov.w = vv.w + pv.w;
    vo[c4] = ov;
  }
}

// per (row, channel): softmax over k neighbours + weighted sum of value
__global__ void __launch_bounds__(256) softmax_agg(
    const float* __restrict__ logits, const float* __restrict__ value,
    bhalf* __restrict__ agg, int rows, int k) {
  const int t = blockIdx.x * blockDim.x + threadIdx.x;
  if (t >= rows * 64) return;
  const int r = t >> 6, c = t & 63;
  const float* lp = logits + (size_t)r * k * 64 + c;
  const float* vp = value + (size_t)r * k * 64 + c;
  float mx = -3e38f;
  for (int j = 0; j < k; ++j) mx = fmaxf(mx, lp[(size_t)j * 64]);
  float s = 0.f, a = 0.f;
  for (int j = 0; j < k; ++j) {
    float e = __expf(lp[(size_t)j * 64] - mx);
    s += e;
    a += e * vp[(size_t)j * 64];
  }
  agg[(size_t)r * 64 + c] = f2b(a / s);
}

// ---------------------------------------------------------------------------
// SA3 (group-all) input build, global max, final output assembly
// ---------------------------------------------------------------------------
__global__ void __launch_bounds__(256) build_sa3(const float* __restrict__ xyz,
                                                 const float* __restrict__ feats,
                                                 bhalf* __restrict__ A, int total) {
  const int r = blockIdx.x * blockDim.x + threadIdx.x;
  if (r >= total) return;
  const float* p = xyz + (size_t)r * 3;
  bhalf* o = A + (size_t)r * 288;
  o[0] = f2b(p[0]); o[1] = f2b(p[1]); o[2] = f2b(p[2]);
  const float* f = feats + (size_t)r * 256;
#pragma unroll 4
  for (int c = 0; c < 256; ++c) o[3 + c] = f2b(f[c]);
  for (int c = 259; c < 288; ++c) o[c] = f2b(0.0f);
}

__global__ void __launch_bounds__(256) max_over_n(const float* __restrict__ h,
                                                  float* __restrict__ out,
                                                  int n, int C, int total) {
  const int t = blockIdx.x * blockDim.x + threadIdx.x;
  if (t >= total) return;
  const int b = t / C, c = t - b * C;
  const float* p = h + (size_t)b * n * C + c;
  float mx = p[0];
  for (int i = 1; i < n; ++i) mx = fmaxf(mx, p[(size_t)i * C]);
  out[t] = mx;
}

__global__ void __launch_bounds__(256) finalize_out(const float* __restrict__ nx2cm,
                                                    const float* __restrict__ l2pts,
                                                    float* __restrict__ out) {
  const int t = blockIdx.x * blockDim.x + threadIdx.x;
  const int total = 6144 + BATCH * 256 * 128;
  if (t >= total) return;
  if (t < 6144) {
    out[16384 + t] = nx2cm[t];                       // l2_xyz (B,3,128)
  } else {
    const int u = t - 6144;                          // l2_points (B,256,128)
    const int b = u / (256 * 128);
    const int r = u - b * 256 * 128;
    const int c = r / 128;
    const int i = r - c * 128;
    out[22528 + u] = l2pts[((size_t)b * 128 + i) * 256 + c];
  }
}

// ---------------------------------------------------------------------------
// Host orchestration
// ---------------------------------------------------------------------------
enum {
  IN_PC = 0, IN_KP,
  SA1_B0, SA1_W0, SA1_B1, SA1_W1,
  SA2_B0, SA2_W0, SA2_B1, SA2_W1,
  SA3_B0, SA3_W0, SA3_B1, SA3_W1,
  TR1_BASE = 14, TR2_BASE = 36,
};
enum {
  T_ATT1_B = 0, T_ATT1_W, T_ATT2_B, T_ATT2_W, T_ATT_B, T_ATT_G,
  T_END_B, T_END_W, T_K_B, T_K_W, T_POS1_B, T_POS1_W,
  T_POS2_B, T_POS2_W, T_POS_B, T_POS_G, T_Q_B, T_Q_W,
  T_START_B, T_START_W, T_V_B, T_V_W,
};

struct TrW { bhalf *start, *q, *kk, *v, *end, *pos2, *att1, *att2; };

extern "C" void kernel_launch(void* const* d_in, const int* in_sizes, int n_in,
                              void* d_out, int out_size, void* d_ws, size_t ws_size,
                              hipStream_t stream) {
  (void)in_sizes; (void)n_in; (void)out_size; (void)ws_size;
  auto P = [&](int i) { return (const float*)d_in[i]; };
  const float* pc = P(IN_PC);
  const float* kp = P(IN_KP);

  char* ws = (char*)d_ws;
  size_t off = 0;
  auto alloc = [&](size_t bytes) -> void* {
    void* p = ws + off;
    off += (bytes + 255) & ~(size_t)255;
    return p;
  };

  // ---- LDS size attributes (>64KB dynamic LDS users) ----
  hipFuncSetAttribute(reinterpret_cast<const void*>(&fps_kernel),
                      hipFuncAttributeMaxDynamicSharedMemorySize, 131072);
  hipFuncSetAttribute(reinterpret_cast<const void*>(&sa_grouped_mlp<128, 160, 128, 256>),
                      hipFuncAttributeMaxDynamicSharedMemorySize, 160 * 1024);
  hipFuncSetAttribute(reinterpret_cast<const void*>(&sa_grouped_mlp<3, 32, 64, 128>),
                      hipFuncAttributeMaxDynamicSharedMemorySize, 64 * 1024);

  // ---- bf16 weight conversion (all GEMM weights, K padded to 32) ----
  auto convw = [&](int idx, int O, int K, int Kp) -> bhalf* {
    bhalf* dst = (bhalf*)alloc((size_t)O * Kp * sizeof(bhalf));
    const int tot = O * Kp;
    convert_pad_w<<<(tot + 255) / 256, 256, 0, stream>>>(P(idx), dst, K, Kp, tot);
    return dst;
  };
  bhalf* w_sa1_1 = convw(SA1_W0, 64, 6, 32);
  bhalf* w_sa1_2 = convw(SA1_W1, 128, 64, 64);
  bhalf* w_sa2_1 = convw(SA2_W0, 128, 131, 160);
  bhalf* w_sa2_2 = convw(SA2_W1, 256, 128, 128);
  bhalf* w_sa3_1 = convw(SA3_W0, 512, 259, 288);
  bhalf* w_sa3_2 = convw(SA3_W1, 1024, 512, 512);
  auto make_trw = [&](int base, int C) {
    TrW w;
    w.att1  = convw(base + T_ATT1_W, 256, 64, 64);
    w.att2  = convw(base + T_ATT2_W, 64, 256, 256);
    w.end   = convw(base + T_END_W, C, 64, 64);
    w.kk    = convw(base + T_K_W, 64, 64, 64);
    w.pos2  = convw(base + T_POS2_W, 64, 64, 64);
    w.q     = convw(base + T_Q_W, 64, 64, 64);
    w.start = convw(base + T_START_W, 64, C, C);
    w.v     = convw(base + T_V_W, 64, 64, 64);
    return w;
  };
  TrW trw1 = make_trw(TR1_BASE, 128);
  TrW trw2 = make_trw(TR2_BASE, 256);

  // ---- activation buffers ----
  float* xyz1  = (float*)alloc((size_t)BATCH * NPTS0 * 3 * 4);
  int*   fps1  = (int*)alloc((size_t)BATCH * S1 * 4);
  float* nx1pm = (float*)alloc((size_t)BATCH * S1 * 3 * 4);
  float* nx1cm = (float*)alloc((size_t)BATCH * S1 * 3 * 4);
  int*   idx1  = (int*)alloc((size_t)BATCH * S1 * 16 * 4);
  float* l1f   = (float*)alloc((size_t)BATCH * S1 * 128 * 4);
  bhalf* l1b16 = (bhalf*)alloc((size_t)BATCH * S1 * 128 * 2);
  float* l1post= (float*)alloc((size_t)BATCH * S1 * 128 * 4);
  int*   fps2  = (int*)alloc((size_t)BATCH * S2 * 4);
  float* nx2pm = (float*)alloc((size_t)BATCH * S2 * 3 * 4);
  float* nx2cm = (float*)alloc((size_t)BATCH * S2 * 3 * 4);
  int*   idx2  = (int*)alloc((size_t)BATCH * S2 * 16 * 4);
  float* l2f   = (float*)alloc((size_t)BATCH * S2 * 256 * 4);
  bhalf* l2b16 = (bhalf*)alloc((size_t)BATCH * S2 * 256 * 2);
  float* l2post= (float*)alloc((size_t)BATCH * S2 * 256 * 4);

  // transformer scratch, sized for tr1 (tr2 reuses; pairs1 > pairs2)
  const int ROWS1 = BATCH * S1;            // 8192
  const int PAIRS1 = ROWS1 * 20;           // 163840
  int*   idxt  = (int*)alloc((size_t)PAIRS1 * 4);
  bhalf* x1b   = (bhalf*)alloc((size_t)ROWS1 * 64 * 2);
  float* qf    = (float*)alloc((size_t)ROWS1 * 64 * 4);
  float* kf    = (float*)alloc((size_t)ROWS1 * 64 * 4);
  float* vf    = (float*)alloc((size_t)ROWS1 * 64 * 4);
  bhalf* hpos  = (bhalf*)alloc((size_t)PAIRS1 * 64 * 2);
  float* pe    = (float*)alloc((size_t)PAIRS1 * 64 * 4);
  bhalf* ain   = (bhalf*)alloc((size_t)PAIRS1 * 64 * 2);
  float* val   = (float*)alloc((size_t)PAIRS1 * 64 * 4);
  bhalf* t1    = (bhalf*)alloc((size_t)PAIRS1 * 256 * 2);
  float* logit = (float*)alloc((size_t)PAIRS1 * 64 * 4);
  bhalf* agg   = (bhalf*)alloc((size_t)ROWS1 * 64 * 2);

  bhalf* a3 = (bhalf*)alloc((size_t)2048 * 288 * 2);
  bhalf* h1 = (bhalf*)alloc((size_t)2048 * 512 * 2);
  float* h2 = (float*)alloc((size_t)2048 * 1024 * 4);

  // gemm launcher: grid (rows/256, N/16), dynamic LDS = 32*Kp bytes (W tile)
  auto gemm = [&](const bhalf* A, const bhalf* W, const float* bias,
                  const float* gamma, const float* beta, const float* resid,
                  float* outF, bhalf* outB, int rows, int Kp, int N, int flags) {
    gemm_rm<<<dim3(rows / 256, N / 16), 256, (size_t)Kp * 32, stream>>>(
        A, W, bias, gamma, beta, resid, outF, outB, Kp, N, flags);
  };

  // ---- stage 0: concat input cloud ----
  {
    const int tot = BATCH * NPTS0;
    concat_xyz<<<(tot + 255) / 256, 256, 0, stream>>>(pc, kp, xyz1);
  }

  // ---- SA1 ----
  fps_kernel<<<BATCH, 1024, (NPTS0 + 2048) * 4, stream>>>(xyz1, NPTS0, S1, fps1);
  gather_xyz<<<(BATCH * S1 + 255) / 256, 256, 0, stream>>>(xyz1, fps1, nx1pm, nx1cm,
                                                           NPTS0, S1);
  knn_kernel<16><<<dim3((S1 + 127) / 128, BATCH), 128, 0, stream>>>(nx1pm, xyz1, S1,
                                                                    NPTS0, idx1);
  sa_grouped_mlp<3, 32, 64, 128>
      <<<dim3(S1 / 16, BATCH), 256, (256 * 32 + 256 * 64) * 2, stream>>>(
          xyz1, xyz1, nx1pm, idx1, w_sa1_1, P(SA1_B0), w_sa1_2, P(SA1_B1),
          l1f, l1b16, NPTS0, S1);

  // ---- vTransformer (shared routine) ----
  auto run_tr = [&](int base, int n, int C, const float* pos_pm,
                    const float* xin_f, const bhalf* xin_b, float* xout_f,
                    const TrW& W) {
    const int rows = BATCH * n;
    const int pairs = rows * 20;
    knn_kernel<20><<<dim3((n + 127) / 128, BATCH), 128, 0, stream>>>(pos_pm, pos_pm,
                                                                     n, n, idxt);
    // start: C -> 64 (bf16 out)
    gemm(xin_b, W.start, P(base + T_START_B), nullptr, nullptr, nullptr,
         nullptr, x1b, rows, C, 64, 0);
    // q, k, v: 64 -> 64 (f32 out)
    gemm(x1b, W.q, P(base + T_Q_B), nullptr, nullptr, nullptr,
         qf, nullptr, rows, 64, 64, 0);
    gemm(x1b, W.kk, P(base + T_K_B), nullptr, nullptr, nullptr,
         kf, nullptr, rows, 64, 64, 0);
    gemm(x1b, W.v, P(base + T_V_B), nullptr, nullptr, nullptr,
         vf, nullptr, rows, 64, 64, 0);
    // positional encoding
    pos_mlp1<<<(pairs + 255) / 256, 256, 0, stream>>>(
        pos_pm, idxt, P(base + T_POS1_W), P(base + T_POS1_B),
        P(base + T_POS_G), P(base + T_POS_B), hpos, n, 20, pairs);
    gemm(hpos, W.pos2, P(base + T_POS2_B), nullptr, nullptr, nullptr,
         pe, nullptr, pairs, 64, 64, 0);
    // attn input + value
    build_attn_value<<<(pairs + 255) / 256, 256, 0, stream>>>(
        qf, kf, vf, pe, idxt, ain, val, n, 20, pairs);
    // att1: 64 -> 256 (BN + relu, bf16 out)
    gemm(ain, W.att1, P(base + T_ATT1_B), P(base + T_ATT_G), P(base + T_ATT_B),
         nullptr, nullptr, t1, pairs, 64, 256, 3);
    // att2: 256 -> 64 (logits f32)
    gemm(t1, W.att2, P(base + T_ATT2_B), nullptr, nullptr, nullptr,
         logit, nullptr, pairs, 256, 64, 0);
    // softmax over 20 neighbours + weighted aggregation
    softmax_agg<<<(rows * 64 + 255) / 256, 256, 0, stream>>>(logit, val, agg,
                                                             rows, 20);
    // end: 64 -> C, + identity residual
    gemm(agg, W.end, P(base + T_END_B), nullptr, nullptr, xin_f,
         xout_f, nullptr, rows, 64, C, 4);
  };
  run_tr(TR1_BASE, S1, 128, nx1pm, l1f, l1b16, l1post, trw1);

  // ---- SA2 ----
  fps_kernel<<<BATCH, 512, (S1 + 1024) * 4, stream>>>(nx1pm, S1, S2, fps2);
  gather_xyz<<<(BATCH * S2 + 255) / 256, 256, 0, stream>>>(nx1pm, fps2, nx2pm, nx2cm,
                                                           S1, S2);
  knn_kernel<16><<<dim3((S2 + 127) / 128, BATCH), 128, 0, stream>>>(nx2pm, nx1pm, S2,
                                                                    S1, idx2);
  sa_grouped_mlp<128, 160, 128, 256>
      <<<dim3(S2 / 16, BATCH), 256, (256 * 160 + 256 * 128) * 2, stream>>>(
          nx1pm, l1post, nx2pm, idx2, w_sa2_1, P(SA2_B0), w_sa2_2, P(SA2_B1),
          l2f, l2b16, S1, S2);

  run_tr(TR2_BASE, S2, 256, nx2pm, l2f, l2b16, l2post, trw2);

  // ---- SA3 (group-all) ----
  build_sa3<<<(2048 + 255) / 256, 256, 0, stream>>>(nx2pm, l2post, a3, 2048);
  gemm(a3, w_sa3_1, P(SA3_B0), nullptr, nullptr, nullptr,
       nullptr, h1, 2048, 288, 512, 1);
  gemm(h1, w_sa3_2, P(SA3_B1), nullptr, nullptr, nullptr,
       h2, nullptr, 2048, 512, 1024, 0);
  max_over_n<<<(BATCH * 1024 + 255) / 256, 256, 0, stream>>>(
      h2, (float*)d_out, 128, 1024, BATCH * 1024);

  // ---- assemble l2_xyz + transposed l2_points ----
  {
    const int tot = 6144 + BATCH * 256 * 128;
    finalize_out<<<(tot + 255) / 256, 256, 0, stream>>>(nx2cm, l2post, (float*)d_out);
  }
}
